// AnemllQATLinearRefactored_6210522710119
// MI455X (gfx1250) — matrix operand order repaired and checked by reading, not compiled, not run on hardware
//
#include <hip/hip_runtime.h>
#include <math.h>

typedef __attribute__((ext_vector_type(4)))  _Float16 v4h;
typedef __attribute__((ext_vector_type(8)))  _Float16 v8h;
typedef __attribute__((ext_vector_type(16))) _Float16 v16h;
typedef __attribute__((ext_vector_type(4)))  float    v4f;
typedef __attribute__((ext_vector_type(8)))  float    v8f;
typedef __attribute__((ext_vector_type(4)))  unsigned int v4u;
typedef __attribute__((ext_vector_type(4)))  int      v4i;
typedef __attribute__((ext_vector_type(8)))  int      v8i;

#define M_DIM 4096   // B*S = 2*2048
#define N_DIM 4096   // D_out
#define K_DIM 4096   // D_in
#define R_DIM 4

#define BLK_M 128
#define BLK_N 256
#define BK    32
#define LDS_STRIDE 40  // 32 halves + 8 pad halves (TDM pad: 4 DWORDs per 16 DWORDs)

// ---------------- prep: Bm = tanh(scale_B), Ag = tanh(scale_A)*g ----------------
__global__ void prep_small_kernel(const float* __restrict__ scale_A,
                                  const float* __restrict__ scale_B,
                                  const float* __restrict__ g,
                                  float* __restrict__ Bm,
                                  float* __restrict__ Ag) {
    int idx = blockIdx.x * blockDim.x + threadIdx.x;
    const int nb = R_DIM * K_DIM;
    const int na = N_DIM * R_DIM;
    if (idx < nb) {
        Bm[idx] = tanhf(scale_B[idx]);
    } else if (idx < nb + na) {
        int j = idx - nb;
        int r = j & (R_DIM - 1);
        Ag[j] = tanhf(scale_A[j]) * g[r];
    }
}

// ---------------- W16[o,i] = f16( Q[o,i] * sum_r Ag[o,r]*Bm[r,i] ) ----------------
__global__ void prep_w_kernel(const float* __restrict__ Q,
                              const float* __restrict__ Bm,
                              const float* __restrict__ Ag,
                              _Float16* __restrict__ W16) {
    long idx4 = (long)blockIdx.x * blockDim.x + threadIdx.x;
    long i4 = idx4 * 4;
    if (i4 >= (long)N_DIM * K_DIM) return;
    int o = (int)(i4 >> 12);
    int i = (int)(i4 & (K_DIM - 1));
    float a0 = Ag[o * R_DIM + 0];
    float a1 = Ag[o * R_DIM + 1];
    float a2 = Ag[o * R_DIM + 2];
    float a3 = Ag[o * R_DIM + 3];
    v4f q = *(const v4f*)(Q + i4);
    v4h w;
#pragma unroll
    for (int t = 0; t < 4; ++t) {
        float e = a0 * Bm[0 * K_DIM + i + t]
                + a1 * Bm[1 * K_DIM + i + t]
                + a2 * Bm[2 * K_DIM + i + t]
                + a3 * Bm[3 * K_DIM + i + t];
        w[t] = (_Float16)(q[t] * e);
    }
    *(v4h*)(W16 + i4) = w;
}

// ---------------- X16 = f16(x) ----------------
__global__ void conv_x_kernel(const float* __restrict__ x, _Float16* __restrict__ X16) {
    long idx4 = (long)blockIdx.x * blockDim.x + threadIdx.x;
    long i4 = idx4 * 4;
    if (i4 >= (long)M_DIM * K_DIM) return;
    v4f v = *(const v4f*)(x + i4);
    v4h h;
#pragma unroll
    for (int t = 0; t < 4; ++t) h[t] = (_Float16)v[t];
    *(v4h*)(X16 + i4) = h;
}

// ---------------- TDM: DMA a rows x 32-half 2D tile (row stride K_DIM) into LDS ----------------
// Pads 4 DWORDs after every 16 DWORDs stored -> LDS row stride = 40 halves.
__device__ __forceinline__ void tdm_load_tile(unsigned lds_off, const _Float16* gptr,
                                              unsigned tile_rows) {
    unsigned long long ga = (unsigned long long)(size_t)gptr;
    v4u g0;
    g0[0] = 1u;                                   // count=1, user descriptor
    g0[1] = lds_off;                              // lds_addr (bytes)
    g0[2] = (unsigned)(ga & 0xFFFFFFFFu);         // global_addr[31:0]
    g0[3] = (unsigned)((ga >> 32) & 0x01FFFFFFu)  // global_addr[56:32]
          | (2u << 30);                           // type = 2 ("image")
    v8i g1;
    g1[0] = (1 << 16)                             // data_size = 2 bytes
          | (1 << 20)                             // pad_enable
          | (3 << 22)                             // pad_interval: 16 DWORDs
          | (3 << 25);                            // pad_amount: 4 DWORDs
    g1[1] = (int)((unsigned)(K_DIM & 0xFFFF) << 16);         // tensor_dim0[15:0]
    g1[2] = (int)(((unsigned)K_DIM >> 16)                    // tensor_dim0[31:16]
          | ((unsigned)(K_DIM & 0xFFFF) << 16));             // tensor_dim1[15:0]
    g1[3] = (int)(((unsigned)K_DIM >> 16)                    // tensor_dim1[31:16]
          | ((unsigned)BK << 16));                           // tile_dim0 = 32
    g1[4] = (int)(tile_rows & 0xFFFF);                       // tile_dim1 ; tile_dim2 = 0
    g1[5] = (int)K_DIM;                                      // tensor_dim0_stride[31:0]
    g1[6] = 0;                                               // stride[47:32] ; dim1_stride lo
    g1[7] = 0;
    v4i z4 = {0, 0, 0, 0};
#if __has_include(<hip/amd_detail/amd_gfx1250_TDM.h>)
    v8i z8 = {0, 0, 0, 0, 0, 0, 0, 0};
    __builtin_amdgcn_tensor_load_to_lds(g0, g1, z4, z4, z8, 0);   // clang-23 6-arg form
#else
    __builtin_amdgcn_tensor_load_to_lds(g0, g1, z4, z4, 0);       // ROCm 7.2 5-arg form
#endif
}

// ---------------- WMMA GEMM: out[m,n] = sum_k X16[m,k]*W16[n,k] + bias[n] ----------------
// Block tile 128x256, 8 waves of 64x64; TDM double-buffered LDS staging.
__global__ __launch_bounds__(256)
void gemm_wmma_kernel(const _Float16* __restrict__ X16,
                      const _Float16* __restrict__ W16,
                      const float* __restrict__ bias,
                      float* __restrict__ out) {
    __shared__ __align__(16) _Float16 lA[2][BLK_M * LDS_STRIDE + 8];
    __shared__ __align__(16) _Float16 lB[2][BLK_N * LDS_STRIDE + 8];

    const int t     = threadIdx.x;
    const int lane  = t & 31;          // wave32
    const int wave  = t >> 5;          // 0..7
    const int waveM = wave >> 2;       // 0..1 -> 64 rows
    const int waveN = wave & 3;        // 0..3 -> 64 cols
    const int mLane = lane & 15;
    const int hi    = lane >> 4;       // 0/1 lane half

    const int blockM = blockIdx.y * BLK_M;
    const int blockN = blockIdx.x * BLK_N;

    const _Float16* gA = X16 + (size_t)blockM * K_DIM;   // tile row base (col k added below)
    const _Float16* gB = W16 + (size_t)blockN * K_DIM;

    const unsigned ldsA0 = (unsigned)(size_t)(void*)&lA[0][0];
    const unsigned ldsA1 = (unsigned)(size_t)(void*)&lA[1][0];
    const unsigned ldsB0 = (unsigned)(size_t)(void*)&lB[0][0];
    const unsigned ldsB1 = (unsigned)(size_t)(void*)&lB[1][0];

    v8f acc[4][4];
    const v8f vzero = {0.f, 0.f, 0.f, 0.f, 0.f, 0.f, 0.f, 0.f};
#pragma unroll
    for (int mf = 0; mf < 4; ++mf)
#pragma unroll
        for (int nf = 0; nf < 4; ++nf)
            acc[mf][nf] = vzero;

    // prologue: DMA first K-slab into buffer 0
    if (wave == 0) {
        tdm_load_tile(ldsA0, gA, BLK_M);
        tdm_load_tile(ldsB0, gB, BLK_N);
        __builtin_amdgcn_s_wait_tensorcnt(0);
    }
    __syncthreads();

    int buf = 0;
    for (int k0 = 0; k0 < K_DIM; k0 += BK) {
        // prefetch next K-slab into the other buffer (overlaps with compute below)
        if (wave == 0 && (k0 + BK) < K_DIM) {
            tdm_load_tile(buf ? ldsA0 : ldsA1, gA + k0 + BK, BLK_M);
            tdm_load_tile(buf ? ldsB0 : ldsB1, gB + k0 + BK, BLK_N);
        }

        const _Float16* cA = &lA[buf][0];
        const _Float16* cB = &lB[buf][0];

        // B fragments: 32x16 (KxN); lane n = mLane, k = hi*16 + 0..15 contiguous
        v16h bfrag[4];
#pragma unroll
        for (int nf = 0; nf < 4; ++nf) {
            const _Float16* pb = cB + (waveN * 64 + nf * 16 + mLane) * LDS_STRIDE + hi * 16;
            v8h blo = *(const v8h*)pb;
            v8h bhi = *(const v8h*)(pb + 8);
            bfrag[nf] = __builtin_shufflevector(blo, bhi,
                0, 1, 2, 3, 4, 5, 6, 7, 8, 9, 10, 11, 12, 13, 14, 15);
        }

        // A fragments: 16x32 (MxK); lane row = mLane, k = hi*16 + 0..15 contiguous
#pragma unroll
        for (int mf = 0; mf < 4; ++mf) {
            const _Float16* pa = cA + (waveM * 64 + mf * 16 + mLane) * LDS_STRIDE + hi * 16;
            v8h alo = *(const v8h*)pa;
            v8h ahi = *(const v8h*)(pa + 8);
            v16h afrag = __builtin_shufflevector(alo, ahi,
                0, 1, 2, 3, 4, 5, 6, 7, 8, 9, 10, 11, 12, 13, 14, 15);
#pragma unroll
            for (int nf = 0; nf < 4; ++nf) {
                acc[mf][nf] = __builtin_amdgcn_wmma_f32_16x16x32_f16(
                    false, afrag, false, bfrag[nf],
                    (short)0, acc[mf][nf], false, false);
            }
        }

        // drain the prefetch DMA, then make it visible to all waves
        if (wave == 0) __builtin_amdgcn_s_wait_tensorcnt(0);
        __syncthreads();
        buf ^= 1;
    }

    // epilogue: C layout VGPR v -> M = v + hi*8, N = mLane
#pragma unroll
    for (int mf = 0; mf < 4; ++mf) {
#pragma unroll
        for (int nf = 0; nf < 4; ++nf) {
            const int n = blockN + waveN * 64 + nf * 16 + mLane;
            const float bv = bias[n];
#pragma unroll
            for (int v = 0; v < 8; ++v) {
                const int m = blockM + waveM * 64 + mf * 16 + hi * 8 + v;
                out[(size_t)m * N_DIM + n] = acc[mf][nf][v] + bv;
            }
        }
    }
}

extern "C" void kernel_launch(void* const* d_in, const int* in_sizes, int n_in,
                              void* d_out, int out_size, void* d_ws, size_t ws_size,
                              hipStream_t stream) {
    const float* x    = (const float*)d_in[0];   // [2,2048,4096]
    const float* wgt  = (const float*)d_in[1];   // [4096,4096]
    const float* bias = (const float*)d_in[2];   // [4096]
    const float* sA   = (const float*)d_in[3];   // [4096,4]
    const float* sB   = (const float*)d_in[4];   // [4,4096]
    const float* g    = (const float*)d_in[5];   // [4]
    float* out = (float*)d_out;

    // ws: Bm f32 [4*4096] | Ag f32 [4096*4] | X16 half [4096*4096] | W16 half [4096*4096]
    float* wsf = (float*)d_ws;
    float* Bm = wsf;
    float* Ag = wsf + R_DIM * K_DIM;
    _Float16* X16 = (_Float16*)(wsf + R_DIM * K_DIM + N_DIM * R_DIM);
    _Float16* W16 = X16 + (size_t)M_DIM * K_DIM;

    const int smallN = R_DIM * K_DIM + N_DIM * R_DIM;
    prep_small_kernel<<<(smallN + 255) / 256, 256, 0, stream>>>(sA, sB, g, Bm, Ag);

    const long wq = (long)N_DIM * K_DIM / 4;
    prep_w_kernel<<<(int)((wq + 255) / 256), 256, 0, stream>>>(wgt, Bm, Ag, W16);

    const long xq = (long)M_DIM * K_DIM / 4;
    conv_x_kernel<<<(int)((xq + 255) / 256), 256, 0, stream>>>(x, X16);

    dim3 grid(N_DIM / BLK_N, M_DIM / BLK_M);   // 16 x 32 workgroups
    gemm_wmma_kernel<<<grid, 256, 0, stream>>>(X16, W16, bias, out);
}
